// EncoderLayer_44805098832331
// MI455X (gfx1250) — matrix operand order repaired and checked by reading, not compiled, not run on hardware
//
#include <hip/hip_runtime.h>
#include <hip/hip_bf16.h>
#include <hip/hip_fp16.h>

// ---------------------------------------------------------------------------
// Transformer encoder layer for MI455X (gfx1250, wave32, WMMA + async-DMA).
// B=8 S=1024 D=1024 H=16 DH=64 F=4096.  ~240 GFLOP of GEMM -> compute bound:
// all matmuls via v_wmma_f32_16x16x32_f16; LDS staging via
// global_load_async_to_lds_b128 (ASYNCcnt) with double-buffered tiles.
// GEMM block tile 128x128 (wave tile 32x64 -> 8 WMMA / 12 ds_load per K-step).
// ---------------------------------------------------------------------------

#define BB 8
#define SS 1024
#define DD 1024
#define HH 16
#define FFDIM 4096
#define DH 64
#define MM (BB * SS)   // 8192 token rows

typedef __attribute__((ext_vector_type(16))) _Float16 v16h;
typedef __attribute__((ext_vector_type(8)))  _Float16 v8h;
typedef __attribute__((ext_vector_type(8)))  float    v8f;

__device__ __forceinline__ v16h cat16(v8h lo, v8h hi) {
  return __builtin_shufflevector(lo, hi, 0,1,2,3,4,5,6,7,8,9,10,11,12,13,14,15);
}

__device__ __forceinline__ v8f wmma_f16(v16h a, v16h b, v8f c) {
  return __builtin_amdgcn_wmma_f32_16x16x32_f16(false, a, false, b, (short)0, c,
                                                false, false);
}

// Async DMA: LDS[lds .. +15] = MEM[gaddr .. +15], tracked by ASYNCcnt.
__device__ __forceinline__ unsigned lds_off(const void* p) {
  return (unsigned)(uintptr_t)p;   // low 32 bits of generic ptr == LDS offset
}
__device__ __forceinline__ void async_load_b128(unsigned lds, const void* gaddr) {
  asm volatile("global_load_async_to_lds_b128 %0, %1, off"
               :: "v"(lds), "v"((unsigned long long)(uintptr_t)gaddr)
               : "memory");
}
__device__ __forceinline__ void wait_async0() {
  asm volatile("s_wait_asynccnt 0x0" ::: "memory");
}

// A fragment (16x32 f16) from LDS [row][k], stride halves (stride*2 % 16 == 0).
__device__ __forceinline__ v16h frag_a(const _Float16* base, int stride, int lane) {
  const _Float16* p = base + (lane & 15) * stride + 8 * (lane >> 4);
  v8h lo = *(const v8h*)p;
  v8h hi = *(const v8h*)(p + 16);
  return cat16(lo, hi);
}

// B fragment (32x16 f16) from LDS stored TRANSPOSED [n][k].
__device__ __forceinline__ v16h frag_b(const _Float16* base, int stride, int lane) {
  const _Float16* p = base + (lane & 15) * stride + 8 * (lane >> 4);
  v8h lo = *(const v8h*)p;
  v8h hi = *(const v8h*)(p + 16);
  return cat16(lo, hi);
}

// ---------------------------------------------------------------------------
// Cast f32 [K][N] weight -> f16 TRANSPOSED [N][K] (one pass, outside hot loop).
__global__ void cast_transpose_kernel(const float* __restrict__ in,
                                      _Float16* __restrict__ out, int K, int N) {
  int i = blockIdx.x * blockDim.x + threadIdx.x;
  if (i < K * N) {
    int k = i / N, n = i - k * N;
    out[(size_t)n * K + k] = (_Float16)in[i];
  }
}

// One block (256 threads = 8 waves) per row; D = 1024.
__global__ __launch_bounds__(256)
void rmsnorm_kernel(const float* __restrict__ x, const float* __restrict__ g,
                    _Float16* __restrict__ h) {
  __shared__ float red[8];
  int row = blockIdx.x;
  const float* xr = x + (size_t)row * DD;
  float vals[4];
  float ss = 0.f;
#pragma unroll
  for (int i = 0; i < 4; ++i) {
    vals[i] = xr[threadIdx.x + 256 * i];
    ss += vals[i] * vals[i];
  }
#pragma unroll
  for (int m = 16; m >= 1; m >>= 1) ss += __shfl_xor(ss, m, 32);
  if ((threadIdx.x & 31) == 0) red[threadIdx.x >> 5] = ss;
  __syncthreads();
  float tot = 0.f;
#pragma unroll
  for (int w = 0; w < 8; ++w) tot += red[w];
  float scale = rsqrtf(tot / (float)DD + 1e-8f);
#pragma unroll
  for (int i = 0; i < 4; ++i) {
    int c = threadIdx.x + 256 * i;
    h[(size_t)row * DD + c] = (_Float16)(vals[i] * scale * g[c]);
  }
}

// ---------------------------------------------------------------------------
// WMMA GEMM: C[M,N] = A[M,K](f16 row-major) * BT[N,K](f16, pre-transposed).
// Block tile 128x128, 8 waves x (32x64) = 2x4 WMMA tiles. Double-buffered LDS,
// async DMA staging. Epilogue: +bias, ReLU, +residual, f32/f16, optional C^T.
// Requires M%128==0, N%128==0, K%32==0.
__global__ __launch_bounds__(256)
void gemm_kernel(const _Float16* __restrict__ A, const _Float16* __restrict__ BT,
                 int M, int N, int K,
                 float* __restrict__ Cf, _Float16* __restrict__ Ch,
                 const float* __restrict__ bias, const float* __restrict__ resid,
                 int relu, int transout) {
  __shared__ _Float16 sA[2][128 * 40];  // [m][k], stride 40 halves (80B)
  __shared__ _Float16 sB[2][128 * 40];  // [n][k]
  int tid  = threadIdx.x;
  int lane = tid & 31;
  int wave = tid >> 5;
  int m0 = blockIdx.y * 128;
  int n0 = blockIdx.x * 128;
  int wm = (wave >> 1) * 32;   // 4 M-groups of 32
  int wn = (wave & 1) * 64;    // 2 N-groups of 64

  // Per-thread DMA chunk coordinates (16B = 8 halves). 128x32 halves = 512
  // chunks per matrix, 2 per thread for A and 2 for B.
  int r0 = tid >> 2,        c0 = (tid & 3) * 8;   // rows 0-63
  int r1 = (tid >> 2) + 64, c1 = c0;              // rows 64-127

  auto issue = [&](int k0, int buf) {
    async_load_b128(lds_off(&sA[buf][r0 * 40 + c0]),
                    A + (size_t)(m0 + r0) * K + k0 + c0);
    async_load_b128(lds_off(&sA[buf][r1 * 40 + c1]),
                    A + (size_t)(m0 + r1) * K + k0 + c1);
    async_load_b128(lds_off(&sB[buf][r0 * 40 + c0]),
                    BT + (size_t)(n0 + r0) * K + k0 + c0);
    async_load_b128(lds_off(&sB[buf][r1 * 40 + c1]),
                    BT + (size_t)(n0 + r1) * K + k0 + c1);
  };

  v8f acc[2][4] = {};
  issue(0, 0);
  int buf = 0;
  for (int k0 = 0; k0 < K; k0 += 32) {
    wait_async0();       // our DMA into sA/sB[buf] done
    __syncthreads();     // all waves' DMA visible; all prior reads retired
    if (k0 + 32 < K) issue(k0 + 32, buf ^ 1);  // stream next tile during WMMAs

    v16h av[2], bv[4];
#pragma unroll
    for (int i = 0; i < 2; ++i)
      av[i] = frag_a(sA[buf] + (wm + 16 * i) * 40, 40, lane);
#pragma unroll
    for (int j = 0; j < 4; ++j)
      bv[j] = frag_b(sB[buf] + (wn + 16 * j) * 40, 40, lane);
#pragma unroll
    for (int i = 0; i < 2; ++i)
#pragma unroll
      for (int j = 0; j < 4; ++j)
        acc[i][j] = wmma_f16(av[i], bv[j], acc[i][j]);
    buf ^= 1;
  }

  // Epilogue. C layout: n = lane&15, m = 8*(lane>>4) + r.
#pragma unroll
  for (int i = 0; i < 2; ++i)
#pragma unroll
    for (int j = 0; j < 4; ++j) {
      int n  = n0 + wn + 16 * j + (lane & 15);
      int mb = m0 + wm + 16 * i + 8 * (lane >> 4);
      float bj = bias ? bias[n] : 0.f;
#pragma unroll
      for (int r = 0; r < 8; ++r) {
        int m = mb + r;
        float v = acc[i][j][r] + bj;
        if (relu) v = fmaxf(v, 0.f);
        if (resid) v += resid[(size_t)m * N + n];
        if (Cf)            Cf[(size_t)m * N + n] = v;
        else if (transout) Ch[(size_t)n * M + m] = (_Float16)v;  // store C^T
        else               Ch[(size_t)m * N + n] = (_Float16)v;
      }
    }
}

// ---------------------------------------------------------------------------
// Flash attention: one block = 64 query rows of one (b,h); 4 waves, 16 q/wave.
// Q,K,O are f16 [B,S,H*DH]; V is pre-transposed f16 VT[D][B*S] so both K and V
// tiles are contiguous async DMA. Online softmax in f32, double-buffered KV.
__global__ __launch_bounds__(128)
void attn_kernel(const _Float16* __restrict__ Q, const _Float16* __restrict__ Kb,
                 const _Float16* __restrict__ VT, const float* __restrict__ relb,
                 const unsigned char* __restrict__ kpm, _Float16* __restrict__ Ob) {
  __shared__ _Float16 sK[2][32 * 72];  // [kv][d]  == B^T layout [n=kv][k=d]
  __shared__ _Float16 sV[2][64 * 40];  // [d][kv]  == B^T layout [n=d][k=kv]
  __shared__ _Float16 sP[64 * 40];     // wave-private P tiles [qrow][kv]
  int tid  = threadIdx.x;
  int lane = tid & 31;
  int wave = tid >> 5;
  int half = lane >> 4;
  int ln   = lane & 15;
  int q0 = blockIdx.x * 64;
  int h  = blockIdx.y;
  int b  = blockIdx.z;
  const size_t hoff = (size_t)h * DH;

  // Per-thread DMA chunk coordinates.
  int rk = tid >> 3, ck = (tid & 7) * 8;   // K: 32 rows x 8 chunks, 2/thread
  int rv = tid >> 2, cv = (tid & 3) * 8;   // V: 64 rows x 4 chunks, 2/thread
  auto issueKV = [&](int kv0, int bufi) {
    async_load_b128(lds_off(&sK[bufi][rk * 72 + ck]),
                    Kb + ((size_t)(b * SS) + kv0 + rk) * DD + hoff + ck);
    async_load_b128(lds_off(&sK[bufi][(rk + 16) * 72 + ck]),
                    Kb + ((size_t)(b * SS) + kv0 + rk + 16) * DD + hoff + ck);
    async_load_b128(lds_off(&sV[bufi][rv * 40 + cv]),
                    VT + (hoff + rv) * MM + (size_t)b * SS + kv0 + cv);
    async_load_b128(lds_off(&sV[bufi][(rv + 32) * 40 + cv]),
                    VT + (hoff + rv + 32) * MM + (size_t)b * SS + kv0 + cv);
  };

  // Preload Q fragments (16 rows x 64 d -> two 16x32 A fragments) from global.
  const _Float16* qp = Q + ((size_t)(b * SS) + q0 + wave * 16 + ln) * DD + hoff;
  v16h qa[2];
#pragma unroll
  for (int c = 0; c < 2; ++c) {
    v8h lo = *(const v8h*)(qp + 32 * c + 8 * half);
    v8h hi = *(const v8h*)(qp + 32 * c + 16 + 8 * half);
    qa[c] = cat16(lo, hi);
  }

  v8f oacc[4] = {};
  float mstat[8], lstat[8];
#pragma unroll
  for (int r = 0; r < 8; ++r) { mstat[r] = -1e30f; lstat[r] = 0.f; }

  issueKV(0, 0);
  int buf = 0;
  for (int kv0 = 0; kv0 < SS; kv0 += 32) {
    wait_async0();
    __syncthreads();
    if (kv0 + 32 < SS) issueKV(kv0 + 32, buf ^ 1);

    // Scores S = (Q K^T)/8 + bias, mask -> two 16x16 f32 tiles per wave.
    v8f sc[2];
#pragma unroll
    for (int t = 0; t < 2; ++t) {
      v8f s = {};
#pragma unroll
      for (int c = 0; c < 2; ++c) {
        const _Float16* p = sK[buf] + (t * 16 + ln) * 72 + 32 * c + 8 * half;
        v16h bf = cat16(*(const v8h*)p, *(const v8h*)(p + 16));
        s = wmma_f16(qa[c], bf, s);
      }
      int kv = kv0 + t * 16 + ln;
      int valid = kpm[b * SS + kv];
      const float* bp =
          relb + ((size_t)h * SS + q0 + wave * 16 + 8 * half) * SS + kv;
#pragma unroll
      for (int r = 0; r < 8; ++r) {
        float v = s[r] * 0.125f + bp[(size_t)r * SS];  // 1/sqrt(64)
        s[r] = valid ? v : -1e30f;
      }
      sc[t] = s;
    }

    // Online softmax per q row (16-lane-group reductions, wave32 shfl).
#pragma unroll
    for (int r = 0; r < 8; ++r) {
      float rmax = fmaxf(sc[0][r], sc[1][r]);
#pragma unroll
      for (int m = 8; m >= 1; m >>= 1) rmax = fmaxf(rmax, __shfl_xor(rmax, m, 32));
      float newm = fmaxf(mstat[r], rmax);
      float p0 = __expf(sc[0][r] - newm);
      float p1 = __expf(sc[1][r] - newm);
      sc[0][r] = p0; sc[1][r] = p1;
      float rsum = p0 + p1;
#pragma unroll
      for (int m = 8; m >= 1; m >>= 1) rsum += __shfl_xor(rsum, m, 32);
      float corr = __expf(mstat[r] - newm);
      lstat[r] = lstat[r] * corr + rsum;
      mstat[r] = newm;
#pragma unroll
      for (int j = 0; j < 4; ++j) oacc[j][r] *= corr;
    }

    // Retile P (C layout -> A layout) via wave-private LDS (same-wave DS order).
#pragma unroll
    for (int t = 0; t < 2; ++t)
#pragma unroll
      for (int r = 0; r < 8; ++r)
        sP[(wave * 16 + 8 * half + r) * 40 + t * 16 + ln] = (_Float16)sc[t][r];

    // O += P(16x32) * V(32x64): 4 WMMAs.
    {
      const _Float16* pp = sP + (wave * 16 + ln) * 40 + 8 * half;
      v16h pf = cat16(*(const v8h*)pp, *(const v8h*)(pp + 16));
#pragma unroll
      for (int j = 0; j < 4; ++j) {
        const _Float16* vp = sV[buf] + (j * 16 + ln) * 40 + 8 * half;
        v16h vf = cat16(*(const v8h*)vp, *(const v8h*)(vp + 16));
        oacc[j] = wmma_f16(pf, vf, oacc[j]);
      }
    }
    buf ^= 1;
  }

  // Normalize by l and store f16 into [B,S,H*DH].
#pragma unroll
  for (int j = 0; j < 4; ++j) {
    int d = j * 16 + ln;
#pragma unroll
    for (int r = 0; r < 8; ++r) {
      int qrow = q0 + wave * 16 + 8 * half + r;
      float v = oacc[j][r] / fmaxf(lstat[r], 1e-20f);
      Ob[((size_t)(b * SS) + qrow) * DD + hoff + d] = (_Float16)v;
    }
  }
}

// ---------------------------------------------------------------------------
extern "C" void kernel_launch(void* const* d_in, const int* in_sizes, int n_in,
                              void* d_out, int out_size, void* d_ws, size_t ws_size,
                              hipStream_t stream) {
  (void)in_sizes; (void)n_in; (void)out_size; (void)ws_size;
  const float* x    = (const float*)d_in[0];
  const unsigned char* kpm = (const unsigned char*)d_in[1];  // bool mask, 1B/elem
  const float* relb = (const float*)d_in[2];
  const float* Wq = (const float*)d_in[3];
  const float* Wk = (const float*)d_in[4];
  const float* Wv = (const float*)d_in[5];
  const float* Wo = (const float*)d_in[6];
  const float* g1 = (const float*)d_in[7];
  const float* g2 = (const float*)d_in[8];
  const float* W1 = (const float*)d_in[9];
  const float* b1 = (const float*)d_in[10];
  const float* W2 = (const float*)d_in[11];
  const float* b2 = (const float*)d_in[12];
  float* out = (float*)d_out;

  char* ws = (char*)d_ws;
  const size_t MB = 1024 * 1024;
  _Float16* wqT  = (_Float16*)(ws +   0 * MB);  //  2 MB  (all weights stored ^T)
  _Float16* wkT  = (_Float16*)(ws +   2 * MB);  //  2 MB
  _Float16* wvT  = (_Float16*)(ws +   4 * MB);  //  2 MB
  _Float16* woT  = (_Float16*)(ws +   6 * MB);  //  2 MB
  _Float16* w1T  = (_Float16*)(ws +   8 * MB);  //  8 MB
  _Float16* w2T  = (_Float16*)(ws +  16 * MB);  //  8 MB
  _Float16* h16  = (_Float16*)(ws +  24 * MB);  // 16 MB (reused as h2 later)
  float*    x2   = (float*)   (ws +  40 * MB);  // 32 MB
  _Float16* q16  = (_Float16*)(ws +  72 * MB);  // 16 MB
  _Float16* k16  = (_Float16*)(ws +  88 * MB);  // 16 MB
  _Float16* vT16 = (_Float16*)(ws + 104 * MB);  // 16 MB  V^T [D][B*S]
  _Float16* att16= (_Float16*)(ws + 120 * MB);  // 16 MB
  _Float16* ff16 = (_Float16*)(ws +  72 * MB);  // 64 MB, reuses q/k/v/att region
  _Float16* h216 = h16;

  // 1) weights -> f16, transposed to [N][K]
  cast_transpose_kernel<<<(DD * DD + 255) / 256, 256, 0, stream>>>(Wq, wqT, DD, DD);
  cast_transpose_kernel<<<(DD * DD + 255) / 256, 256, 0, stream>>>(Wk, wkT, DD, DD);
  cast_transpose_kernel<<<(DD * DD + 255) / 256, 256, 0, stream>>>(Wv, wvT, DD, DD);
  cast_transpose_kernel<<<(DD * DD + 255) / 256, 256, 0, stream>>>(Wo, woT, DD, DD);
  cast_transpose_kernel<<<(DD * FFDIM + 255) / 256, 256, 0, stream>>>(W1, w1T, DD, FFDIM);
  cast_transpose_kernel<<<(FFDIM * DD + 255) / 256, 256, 0, stream>>>(W2, w2T, FFDIM, DD);

  // 2) h = rmsnorm(x, g1) -> f16
  rmsnorm_kernel<<<MM, 256, 0, stream>>>(x, g1, h16);

  // 3) Q/K projections ([B,S,H*DH] f16); V projection stored transposed.
  dim3 gD(DD / 128, MM / 128);
  gemm_kernel<<<gD, 256, 0, stream>>>(h16, wqT, MM, DD, DD, nullptr, q16,
                                      nullptr, nullptr, 0, 0);
  gemm_kernel<<<gD, 256, 0, stream>>>(h16, wkT, MM, DD, DD, nullptr, k16,
                                      nullptr, nullptr, 0, 0);
  gemm_kernel<<<gD, 256, 0, stream>>>(h16, wvT, MM, DD, DD, nullptr, vT16,
                                      nullptr, nullptr, 0, 1);

  // 4) attention (flash, WMMA, async KV streaming) -> att16 f16
  dim3 gA(SS / 64, HH, BB);
  attn_kernel<<<gA, 128, 0, stream>>>(q16, k16, vT16, relb, kpm, att16);

  // 5) x2 = x + att @ Wo   (f32)
  gemm_kernel<<<gD, 256, 0, stream>>>(att16, woT, MM, DD, DD, x2, nullptr,
                                      nullptr, x, 0, 0);

  // 6) h2 = rmsnorm(x2, g2) -> f16
  rmsnorm_kernel<<<MM, 256, 0, stream>>>(x2, g2, h216);

  // 7) ff = relu(h2 @ W1 + b1) -> f16
  dim3 gF(FFDIM / 128, MM / 128);
  gemm_kernel<<<gF, 256, 0, stream>>>(h216, w1T, MM, FFDIM, DD, nullptr, ff16,
                                      b1, nullptr, 1, 0);

  // 8) out = x2 + ff @ W2 + b2  (f32)
  gemm_kernel<<<gD, 256, 0, stream>>>(ff16, w2T, MM, DD, FFDIM, out, nullptr,
                                      b2, x2, 0, 0);
}
